// VQQuantizer_80324478370151
// MI455X (gfx1250) — matrix operand order repaired and checked by reading, not compiled, not run on hardware
//
#include <hip/hip_runtime.h>
#include <cstdint>
#include <cstddef>

// Problem constants (match reference)
#define BB 16
#define DD 512
#define TT 2048
#define KK 8
#define CC 1024
#define CHD 64            // D / K
#define COMMIT 0.25f
#define EPSV 1e-8f

#define QN ((size_t)BB * DD * TT)         // 16,777,216 quantized elements
#define NQT 4                             // t-tiles (of 16) per wave -> N=64
#define TSPAN (16 * NQT)                  // 64 t per wave
#define TILES_PER_K (TT / TSPAN)          // 32
#define WAVES_TOTAL (BB * KK * TILES_PER_K) // 4096

typedef __attribute__((ext_vector_type(16))) __bf16 v16bf;
typedef __attribute__((ext_vector_type(8)))  float  v8f;

// ---------------------------------------------------------------------------
// Kernel 1: prep — zero counts, convert codebook fp32 -> bf16 (row-major
// [k][c][ch]), compute -0.5*||c||^2 per code.
// ---------------------------------------------------------------------------
__global__ __launch_bounds__(256)
void vq_prep_kernel(const float* __restrict__ cb,
                    __bf16* __restrict__ cbb,
                    float* __restrict__ cnorm,
                    unsigned int* __restrict__ counts) {
    int i = blockIdx.x * 256 + threadIdx.x;   // 0 .. K*C-1
    if (i >= KK * CC) return;
    counts[i] = 0u;
    const float* row = cb + (size_t)i * CHD;
    __bf16* orow = cbb + (size_t)i * CHD;
    float s = 0.0f;
#pragma unroll
    for (int ch = 0; ch < CHD; ch += 4) {
        float4 v = *(const float4*)(row + ch);
        s = fmaf(v.x, v.x, s);
        s = fmaf(v.y, v.y, s);
        s = fmaf(v.z, v.z, s);
        s = fmaf(v.w, v.w, s);
        orow[ch + 0] = static_cast<__bf16>(v.x);
        orow[ch + 1] = static_cast<__bf16>(v.y);
        orow[ch + 2] = static_cast<__bf16>(v.z);
        orow[ch + 3] = static_cast<__bf16>(v.w);
    }
    cnorm[i] = -0.5f * s;
}

// ---------------------------------------------------------------------------
// Kernel 2: main — one wave per (b, k, 64-wide t-span).
// A = codebook tile (16 codes x 64 ch, bf16) reused across NQT=4 t-tiles,
// B = x tiles (64 ch x 16 t each, bf16, loop-invariant in VGPRs),
// C init = -0.5*||c||^2 -> WMMA result = dot - 0.5*||c||^2 (argmax == argmin
// of distance). 64 code-tiles x 8 WMMAs, per-lane running argmax, then gather
// fp32 code rows, write quantized output, SSE partials, usage counts.
// ---------------------------------------------------------------------------
__global__ __launch_bounds__(256)
void vq_main_kernel(const float* __restrict__ x,
                    const float* __restrict__ cb,
                    const __bf16* __restrict__ cbb,
                    const float* __restrict__ cnorm,
                    unsigned int* __restrict__ counts,
                    float* __restrict__ partials,
                    float* __restrict__ out) {
    const int wave = threadIdx.x >> 5;
    const int lane = threadIdx.x & 31;
    const int n    = lane & 15;       // N index (t within tile) / A row (code)
    const int hi   = lane >> 4;       // half select

    const int gw  = blockIdx.x * 8 + wave;        // 0 .. 4095
    const int b   = gw >> 8;                      // / (K * 32)
    const int rem = gw & 255;
    const int k   = rem >> 5;                     // / 32
    const int t0  = (rem & 31) << 6;              // * 64

    // ---- Load B (x tiles) once: bf16, ISA B-layout (lane n = column t,
    //      element e -> ch = step*32 + hi*8 + e + (e & 8)). 4 t-tiles x 2 K-steps.
    const float* xk = x + ((size_t)b * DD + (size_t)k * CHD) * TT + (size_t)(t0 + n);
    union BV { v16bf v; __bf16 e[16]; } bx[NQT][2];
#pragma unroll
    for (int q = 0; q < NQT; ++q) {
#pragma unroll
        for (int e = 0; e < 16; ++e) {
            bx[q][0].e[e] = static_cast<__bf16>(xk[(size_t)(hi * 8 + e + (e & 8)) * TT + q * 16]);
            bx[q][1].e[e] = static_cast<__bf16>(xk[(size_t)(32 + hi * 8 + e + (e & 8)) * TT + q * 16]);
        }
    }

    // ---- Sweep 64 code tiles, per t-tile running argmax of (dot - 0.5||c||^2).
    float bs[NQT];
    int   bi[NQT];
#pragma unroll
    for (int q = 0; q < NQT; ++q) { bs[q] = -3.0e38f; bi[q] = 0; }

    const __bf16* arow0 = cbb + ((size_t)k * CC + (size_t)n) * CHD + 8 * hi;
    const float*  cnk   = cnorm + k * CC + 8 * hi;

#pragma unroll 2
    for (int tile = 0; tile < 64; ++tile) {
        const __bf16* row = arow0 + (size_t)tile * 16 * CHD;
        __builtin_prefetch(row + 16 * CHD, 0, 1);   // next code tile (L2 stream)
        union AV { v16bf v; uint4 u[2]; } a0, a1;
        a0.u[0] = *(const uint4*)(row);           // ch  8*hi      .. +7
        a0.u[1] = *(const uint4*)(row + 16);      // ch 16+8*hi    .. +7
        a1.u[0] = *(const uint4*)(row + 32);      // ch 32+8*hi    .. +7
        a1.u[1] = *(const uint4*)(row + 48);      // ch 48+8*hi    .. +7

        const float4* cn4 = (const float4*)(cnk + tile * 16);
        float4 c0 = cn4[0], c1 = cn4[1];

        const int cb0 = (tile << 4) + (hi << 3);  // code index of acc[0]
#pragma unroll
        for (int q = 0; q < NQT; ++q) {
            v8f acc = { c0.x, c0.y, c0.z, c0.w, c1.x, c1.y, c1.z, c1.w };
            acc = __builtin_amdgcn_wmma_f32_16x16x32_bf16(
                      false, a0.v, false, bx[q][0].v, (short)0, acc, false, false);
            acc = __builtin_amdgcn_wmma_f32_16x16x32_bf16(
                      false, a1.v, false, bx[q][1].v, (short)0, acc, false, false);
#pragma unroll
            for (int j = 0; j < 8; ++j) {
                float sc = acc[j];
                if (sc > bs[q]) { bs[q] = sc; bi[q] = cb0 + j; }
            }
        }
    }

    // ---- Per t-tile: merge M-halves, bump counts, write quantized + SSE.
    float sse = 0.0f;
#pragma unroll
    for (int q = 0; q < NQT; ++q) {
        float os = __shfl_xor(bs[q], 16);
        int   oi = __shfl_xor(bi[q], 16);
        int   sel = bi[q];
        if (os > bs[q] || (os == bs[q] && oi < sel)) sel = oi;

        if (hi == 0) atomicAdd(&counts[k * CC + sel], 1u);

        const float* qrow = cb + ((size_t)k * CC + (size_t)sel) * CHD;
        const size_t obase = ((size_t)b * DD + (size_t)k * CHD) * TT
                           + (size_t)(t0 + q * 16 + n);
#pragma unroll 8
        for (int it = 0; it < 32; ++it) {
            int ch = (hi << 5) + it;
            float qv = qrow[ch];
            size_t off = obase + (size_t)ch * TT;
            float xv = x[off];
            out[off] = qv;
            float d = xv - qv;
            sse = fmaf(d, d, sse);
        }
    }
#pragma unroll
    for (int m = 1; m < 32; m <<= 1) sse += __shfl_xor(sse, m);
    if (lane == 0) partials[gw] = sse;
}

// ---------------------------------------------------------------------------
// Kernel 3: finalize — deterministic loss reduction + perplexity per k.
// ---------------------------------------------------------------------------
__global__ __launch_bounds__(256)
void vq_finalize_kernel(const float* __restrict__ partials,
                        const unsigned int* __restrict__ counts,
                        float* __restrict__ out) {
    __shared__ float red[256];
    float s = 0.0f;
    for (int j = 0; j < WAVES_TOTAL / 256; ++j)
        s += partials[threadIdx.x + 256 * j];
    red[threadIdx.x] = s;
    __syncthreads();
    for (int st = 128; st > 0; st >>= 1) {
        if ((int)threadIdx.x < st) red[threadIdx.x] += red[threadIdx.x + st];
        __syncthreads();
    }
    if (threadIdx.x == 0)
        out[QN] = (red[0] / (float)QN) * (1.0f + COMMIT);

    // perplexity: one wave per k
    const int w = threadIdx.x >> 5, lane = threadIdx.x & 31;
    if (w < KK) {
        const float inv = 1.0f / (float)(BB * TT);
        float h = 0.0f;
        for (int c = lane; c < CC; c += 32) {
            float p = (float)counts[w * CC + c] * inv;
            h -= p * __logf(p + EPSV);
        }
#pragma unroll
        for (int m = 1; m < 32; m <<= 1) h += __shfl_xor(h, m);
        if (lane == 0) out[QN + 1 + w] = __expf(h);
    }
}

// ---------------------------------------------------------------------------
// Launch
// ---------------------------------------------------------------------------
extern "C" void kernel_launch(void* const* d_in, const int* in_sizes, int n_in,
                              void* d_out, int out_size, void* d_ws, size_t ws_size,
                              hipStream_t stream) {
    (void)in_sizes; (void)n_in; (void)out_size; (void)ws_size;
    const float* x  = (const float*)d_in[0];       // (B, D, T) fp32
    const float* cb = (const float*)d_in[1];       // (K, C, CH) fp32
    float* out = (float*)d_out;                    // quantized | loss | perplexity[K]

    char* ws = (char*)d_ws;
    const size_t off_cbb    = 0;                                     // 1 MiB bf16 codebook
    const size_t off_cnorm  = off_cbb   + (size_t)KK * CC * CHD * 2; // 32 KiB
    const size_t off_counts = off_cnorm + (size_t)KK * CC * 4;       // 32 KiB
    const size_t off_part   = off_counts + (size_t)KK * CC * 4;      // 16 KiB
    __bf16*       cbb      = (__bf16*)(ws + off_cbb);
    float*        cnorm    = (float*)(ws + off_cnorm);
    unsigned int* counts   = (unsigned int*)(ws + off_counts);
    float*        partials = (float*)(ws + off_part);

    vq_prep_kernel<<<(KK * CC + 255) / 256, 256, 0, stream>>>(cb, cbb, cnorm, counts);
    vq_main_kernel<<<WAVES_TOTAL / 8, 256, 0, stream>>>(x, cb, cbb, cnorm, counts,
                                                        partials, out);
    vq_finalize_kernel<<<1, 256, 0, stream>>>(partials, counts, out);
}
